// CorrelationBranch_61847529063041
// MI455X (gfx1250) — matrix operand order repaired and checked by reading, not compiled, not run on hardware
//
#include <hip/hip_runtime.h>
#include <hip/hip_bf16.h>
#include <math.h>

typedef __attribute__((ext_vector_type(16))) __bf16 v16bf;
typedef __attribute__((ext_vector_type(8)))  float  v8f;

// ---------------------------------------------------------------------------
// Kernel 1: v[b][m][n] = sum_k W[m][k] * x_vit[b][k][n] + bias[m]
//   M = 256 out channels, K = 512 in channels, N = 1024 pixels (32x32), b = 8.
//   One wave (32 lanes) computes one 16x16 tile of D via bf16 WMMA with
//   hi/lo error compensation (3 WMMAs per 16x16x32 K-step, f32 accumulate).
// ---------------------------------------------------------------------------
__global__ __launch_bounds__(32) void corr_gemm_wmma(
    const float* __restrict__ Wm,      // [256][512]
    const float* __restrict__ xv,      // [8][512][1024]
    const float* __restrict__ bias,    // [256]
    float* __restrict__ vout)          // [8][256][1024]
{
  const int lane   = threadIdx.x;      // 0..31
  const int hi     = lane >> 4;        // 0 or 1 (lane half)
  const int lane16 = lane & 15;
  const int n0 = blockIdx.x * 16;      // pixel tile
  const int m0 = blockIdx.y * 16;      // out-channel tile
  const int b  = blockIdx.z;

  const float* xvb = xv + (size_t)b * 512u * 1024u;

  v8f acc = {};
  const int Mrow = m0 + lane16;        // A-matrix row held by this lane
  const int Ncol = n0 + lane16;        // B/D column held by this lane

  for (int k0 = 0; k0 < 512; k0 += 32) {
    // ---- A fragment (W): ISA layout for 16-bit A 16x32:
    //   element e=0..7  -> K = e      + 8*hi
    //   element e=8..15 -> K = (e-8) + 16 + 8*hi
    // => two contiguous 8-float groups per lane.
    float a[16];
    const float* wrow = Wm + (size_t)Mrow * 512u + (size_t)(k0 + hi * 8);
    #pragma unroll
    for (int e = 0; e < 8; ++e) a[e]     = wrow[e];
    #pragma unroll
    for (int e = 0; e < 8; ++e) a[8 + e] = wrow[16 + e];

    // ---- B fragment (x_vit): same K layout as A, column Ncol.
    float bv[16];
    const float* bcol = xvb + (size_t)(k0 + 8 * hi) * 1024u + Ncol;
    #pragma unroll
    for (int e = 0; e < 16; ++e) bv[e] = bcol[(size_t)(e + (e >= 8 ? 8 : 0)) * 1024u];

    // ---- hi/lo bf16 split (a = ah + al to ~16 mantissa bits)
    v16bf Ah, Al, Bh, Bl;
    #pragma unroll
    for (int e = 0; e < 16; ++e) {
      __bf16 ah = (__bf16)a[e];
      Ah[e] = ah;
      Al[e] = (__bf16)(a[e] - (float)ah);
      __bf16 bh = (__bf16)bv[e];
      Bh[e] = bh;
      Bl[e] = (__bf16)(bv[e] - (float)bh);
    }

    // D += Ah*Bh + Ah*Bl + Al*Bh  (al*bl term ~2^-18, negligible)
    acc = __builtin_amdgcn_wmma_f32_16x16x32_bf16(false, Ah, false, Bh,
                                                  (short)0, acc, false, false);
    acc = __builtin_amdgcn_wmma_f32_16x16x32_bf16(false, Ah, false, Bl,
                                                  (short)0, acc, false, false);
    acc = __builtin_amdgcn_wmma_f32_16x16x32_bf16(false, Al, false, Bh,
                                                  (short)0, acc, false, false);
  }

  // ---- store D + bias; D VGPR r: M = m0 + r + 8*hi, N = Ncol
  #pragma unroll
  for (int r = 0; r < 8; ++r) {
    const int m = m0 + r + 8 * hi;
    vout[((size_t)b * 256u + (size_t)m) * 1024u + Ncol] = acc[r] + bias[m];
  }
}

// ---------------------------------------------------------------------------
// Kernel 2: fused bilinear upsample (32x32 -> 128x128, half-pixel centers)
//           + per-pixel cosine similarity over 256 channels.
//   Streams x_cnn (128 MiB) once; v (8 MiB) is L2-resident.
// ---------------------------------------------------------------------------
__global__ __launch_bounds__(256) void corr_fused_cosine(
    const float* __restrict__ xc,      // [8][256][128][128]
    const float* __restrict__ v,       // [8][256][32][32] (flattened 32x32=1024)
    float* __restrict__ out)           // [8][1][128][128]
{
  const int idx = blockIdx.x * 256 + threadIdx.x;  // 0 .. 131071
  const int w = idx & 127;
  const int h = (idx >> 7) & 127;
  const int b = idx >> 14;

  // half-pixel mapping, scale 0.25
  const float sy = (h + 0.5f) * 0.25f - 0.5f;
  const float sx = (w + 0.5f) * 0.25f - 0.5f;
  const float fy0 = floorf(sy), fx0 = floorf(sx);
  const float ty = sy - fy0, tx = sx - fx0;
  const int y0 = min(31, max(0, (int)fy0));
  const int y1 = min(31, max(0, (int)fy0 + 1));
  const int x0 = min(31, max(0, (int)fx0));
  const int x1 = min(31, max(0, (int)fx0 + 1));
  const float w00 = (1.0f - ty) * (1.0f - tx);
  const float w01 = (1.0f - ty) * tx;
  const float w10 = ty * (1.0f - tx);
  const float w11 = ty * tx;

  const int o00 = y0 * 32 + x0, o01 = y0 * 32 + x1;
  const int o10 = y1 * 32 + x0, o11 = y1 * 32 + x1;

  const float* vb = v  + (size_t)b * 256u * 1024u;
  const float* xb = xc + (size_t)b * 256u * 16384u + (size_t)h * 128u + w;

  float dot = 0.0f, nx = 0.0f, nv = 0.0f;
  #pragma unroll 4
  for (int c = 0; c < 256; ++c) {
    const float* vc = vb + (size_t)c * 1024u;
    const float t = w00 * vc[o00] + w01 * vc[o01] +
                    w10 * vc[o10] + w11 * vc[o11];
    const float x = xb[(size_t)c * 16384u];
    dot = fmaf(x, t, dot);
    nx  = fmaf(x, x, nx);
    nv  = fmaf(t, t, nv);
  }
  out[idx] = dot / (sqrtf(nx) * sqrtf(nv) + 1e-8f);
}

// ---------------------------------------------------------------------------
extern "C" void kernel_launch(void* const* d_in, const int* in_sizes, int n_in,
                              void* d_out, int out_size, void* d_ws, size_t ws_size,
                              hipStream_t stream) {
  (void)in_sizes; (void)n_in; (void)out_size; (void)ws_size;
  const float* x_cnn = (const float*)d_in[0];  // [8,256,128,128]
  const float* x_vit = (const float*)d_in[1];  // [8,512,32,32]
  const float* Wm    = (const float*)d_in[2];  // [256,512]
  const float* bias  = (const float*)d_in[3];  // [256]
  float* v = (float*)d_ws;                     // scratch: 8*256*1024 f32 = 8 MiB

  // GEMM: 64 pixel-tiles x 16 channel-tiles x 8 batches, 1 wave each
  dim3 g1(64, 16, 8);
  corr_gemm_wmma<<<g1, 32, 0, stream>>>(Wm, x_vit, bias, v);

  // Fused upsample + cosine: 131072 pixels / 256 threads
  corr_fused_cosine<<<512, 256, 0, stream>>>(x_cnn, v, (float*)d_out);
}
